// Conv3D_Aware_CrossAttention_55190329754001
// MI455X (gfx1250) — matrix operand order repaired and checked by reading, not compiled, not run on hardware
//
#include <hip/hip_runtime.h>
#include <hip/hip_bf16.h>

typedef __bf16 bf16_t;
typedef bf16_t v16bf __attribute__((ext_vector_type(16)));
typedef bf16_t v8bf  __attribute__((ext_vector_type(8)));
typedef bf16_t v4bf  __attribute__((ext_vector_type(4)));
typedef float  v8f   __attribute__((ext_vector_type(8)));
typedef unsigned int u32x4 __attribute__((ext_vector_type(4)));
typedef int i32x8 __attribute__((ext_vector_type(8)));
typedef int i32x4 __attribute__((ext_vector_type(4)));

#define DEV static __device__ __forceinline__

DEV bf16_t f2bf(float f) { return (bf16_t)f; }

DEV v8f vzero8() {
    v8f z;
#pragma unroll
    for (int t = 0; t < 8; ++t) z[t] = 0.0f;
    return z;
}

// D = A(16x32) * B(32x16) + C, bf16 inputs, f32 accum
DEV v8f wmma_bf16(v16bf a, v16bf b, v8f c) {
    return __builtin_amdgcn_wmma_f32_16x16x32_bf16(
        /*neg_a=*/false, a, /*neg_b=*/false, b,
        /*c_mod=*/(short)0, c, /*reuse_a=*/false, /*reuse_b=*/false);
}

// A-matrix fragment (16x32 bf16), row-major source, ld = row stride (elements).
// lanes 0-15 row M=lane hold K=0..15; lanes 16-31 row M=lane-16 hold
// K=16..31.  Each half = 8 contiguous bf16 = one 16B load.
DEV v16bf load_a_bf16(const bf16_t* __restrict__ base, int row0, long ld, int k0) {
    const int lane = threadIdx.x & 31;
    const int r    = lane & 15;
    const int klo  = (lane >> 4) * 16;
    const bf16_t* p = base + (long)(row0 + r) * ld + k0 + klo;
    v8bf lo = *(const v8bf*)(p);
    v8bf hi = *(const v8bf*)(p + 8);
    v16bf f;
#pragma unroll
    for (int t = 0; t < 8; ++t) { f[t] = lo[t]; f[t + 8] = hi[t]; }
    return f;
}

// B-matrix fragment for NT GEMM: out[m,n] = sum_k A[m,k]*W[n,k].
// lane reads 16 contiguous bf16 from row (n0+n) of W (K=0..15 lanes 0-15,
// K=16..31 lanes 16-31).
DEV v16bf load_b_bf16(const bf16_t* __restrict__ base, int n0, long ld, int k0) {
    const int lane = threadIdx.x & 31;
    const int n    = lane & 15;
    const int klo  = (lane >> 4) * 16;
    const bf16_t* p = base + (long)(n0 + n) * ld + k0 + klo;
    v8bf lo = *(const v8bf*)(p);
    v8bf hi = *(const v8bf*)(p + 8);
    v16bf f;
#pragma unroll
    for (int t = 0; t < 8; ++t) { f[t] = lo[t]; f[t + 8] = hi[t]; }
    return f;
}

// Same B-fragment but from an LDS-resident 64x64 tile (ld = 64), filled by TDM.
DEV v16bf load_b_lds(const bf16_t* base, int n0, int k0) {
    const int lane = threadIdx.x & 31;
    const int n    = lane & 15;
    const int klo  = (lane >> 4) * 16;
    const bf16_t* p = base + (n0 + n) * 64 + k0 + klo;
    v8bf lo = *(const v8bf*)(p);
    v8bf hi = *(const v8bf*)(p + 8);
    v16bf f;
#pragma unroll
    for (int t = 0; t < 8; ++t) { f[t] = lo[t]; f[t + 8] = hi[t]; }
    return f;
}

// Issue a 1-D TDM copy of 8KB (4096 bf16) from contiguous global to LDS.
// D# per CDNA5 ISA ch.8: group0 = {count=1, lds_addr, global_addr, type=2};
// group1 = {data_size=2B, tensor_dim0=4096, tensor_dim1=1, tile_dim0=4096,
//           tile_dim1=0 (1-D), stride0=4096}; groups 2/3 zero (2-D form).
DEV void tdm_load_8kb(const bf16_t* gsrc, unsigned lds_byte_off) {
    const unsigned long long ga = (unsigned long long)(const void*)gsrc;
    u32x4 g0;
    g0[0] = 1u;                                           // count=1
    g0[1] = lds_byte_off;                                 // lds_addr
    g0[2] = (unsigned)ga;                                 // global_addr[31:0]
    g0[3] = (unsigned)((ga >> 32) & 0x01FFFFFFu) | (2u << 30);  // addr hi | type=2
    i32x8 g1;
    g1[0] = (int)(1u << 16);       // workgroup_mask=0, data_size=1 -> 2 bytes
    g1[1] = (int)(4096u << 16);    // tensor_dim0[15:0]=4096 (bits 63:48)
    g1[2] = (int)(1u << 16);       // tensor_dim0[31:16]=0, tensor_dim1[15:0]=1
    g1[3] = (int)(4096u << 16);    // tensor_dim1[31:16]=0, tile_dim0=4096
    g1[4] = 0;                     // tile_dim1=0, tile_dim2=0
    g1[5] = 4096;                  // tensor_dim0_stride[31:0]
    g1[6] = 0;
    g1[7] = 0;
    i32x4 gz4;
    gz4[0] = 0; gz4[1] = 0; gz4[2] = 0; gz4[3] = 0;
    i32x8 gz8;
#pragma unroll
    for (int t = 0; t < 8; ++t) gz8[t] = 0;
    __builtin_amdgcn_tensor_load_to_lds(g0, g1, gz4, gz4, gz8, 0);
}

// 64x64 output tile, K=64, bf16 A (lda) x bf16 B (ldb), f32 accumulate.
DEV void gemm_64x64_k64(const bf16_t* __restrict__ A, long lda,
                        const bf16_t* __restrict__ B, long ldb,
                        v8f acc[4][4]) {
#pragma unroll
    for (int k0 = 0; k0 < 64; k0 += 32) {
        v16bf af[4], bfr[4];
#pragma unroll
        for (int i = 0; i < 4; ++i) af[i] = load_a_bf16(A, 16 * i, lda, k0);
#pragma unroll
        for (int j = 0; j < 4; ++j) bfr[j] = load_b_bf16(B, 16 * j, ldb, k0);
#pragma unroll
        for (int i = 0; i < 4; ++i)
#pragma unroll
            for (int j = 0; j < 4; ++j)
                acc[i][j] = wmma_bf16(af[i], bfr[j], acc[i][j]);
    }
}

// ---------------------------------------------------------------------------
// Kernel 0a: convert the four 1024x1024 fp32 weights to bf16 (z = which)
// ---------------------------------------------------------------------------
__global__ __launch_bounds__(256) void k_cvt_w(const float* __restrict__ wq,
                                               const float* __restrict__ wk,
                                               const float* __restrict__ wv,
                                               const float* __restrict__ wp,
                                               bf16_t* __restrict__ Wb) {
    const int z = blockIdx.z;
    const float* s = (z == 0) ? wq : (z == 1) ? wk : (z == 2) ? wv : wp;
    const long idx = (long)blockIdx.x * 256 + threadIdx.x;
    Wb[(long)z * 1048576 + idx] = f2bf(s[idx]);
}

// ---------------------------------------------------------------------------
// Kernel 0b: bulk fp32 -> bf16 (4 elems/thread), used for x and O
// ---------------------------------------------------------------------------
__global__ __launch_bounds__(256) void k_cvt4(const float* __restrict__ src,
                                              bf16_t* __restrict__ dst) {
    const long i4 = ((long)blockIdx.x * 256 + threadIdx.x) * 4;
    const float4 v = *(const float4*)(src + i4);
    v4bf o;
    o[0] = f2bf(v.x); o[1] = f2bf(v.y); o[2] = f2bf(v.z); o[3] = f2bf(v.w);
    *(v4bf*)(dst + i4) = o;
}

// ---------------------------------------------------------------------------
// Kernel 1: fused QKV projection from bf16 x.  M=98304, K=1024, N=1024.
// grid = (Nblk=8, Mblk=384, z) so N-blocks sharing A rows are adjacent (L2).
//   Q -> natural bf16 layout, pre-scaled by d^-0.5
//   K -> natural bf16 layout
//   V -> transposed layouts Vt1[b,g,a,h,dd,j], Vt2[b,g,j,h,dd,a]
// ---------------------------------------------------------------------------
__global__ __launch_bounds__(256) void k_qkv(const bf16_t* __restrict__ Xb,
                                             const bf16_t* __restrict__ Wb,
                                             bf16_t* __restrict__ Qb,
                                             bf16_t* __restrict__ Kb,
                                             bf16_t* __restrict__ Vt1,
                                             bf16_t* __restrict__ Vt2) {
    const int wid  = threadIdx.x >> 5;
    const int lane = threadIdx.x & 31;
    const int z    = blockIdx.z;
    const int m0   = blockIdx.y * 256 + (wid & 3) * 64;
    const int n0   = blockIdx.x * 128 + (wid >> 2) * 64;
    const bf16_t* W = Wb + (long)z * 1048576;

    v8f acc[4][4];
#pragma unroll
    for (int i = 0; i < 4; ++i)
#pragma unroll
        for (int j = 0; j < 4; ++j) acc[i][j] = vzero8();

    for (int k0 = 0; k0 < 1024; k0 += 32) {
        if (k0 + 32 < 1024)
            __builtin_prefetch(Xb + (long)(m0 + (lane & 15)) * 1024 + k0 + 32, 0, 1);
        v16bf af[4], bfr[4];
#pragma unroll
        for (int i = 0; i < 4; ++i) af[i] = load_a_bf16(Xb, m0 + 16 * i, 1024, k0);
#pragma unroll
        for (int j = 0; j < 4; ++j) bfr[j] = load_b_bf16(W, n0 + 16 * j, 1024, k0);
#pragma unroll
        for (int i = 0; i < 4; ++i)
#pragma unroll
            for (int j = 0; j < 4; ++j)
                acc[i][j] = wmma_bf16(af[i], bfr[j], acc[i][j]);
    }

    const int cn = lane & 15;
    const int rb = (lane >> 4) << 3;
    if (z == 0) {
#pragma unroll
        for (int i = 0; i < 4; ++i)
#pragma unroll
            for (int j = 0; j < 4; ++j)
#pragma unroll
                for (int r = 0; r < 8; ++r) {
                    const int row = m0 + 16 * i + rb + r;
                    const int col = n0 + 16 * j + cn;
                    Qb[(long)row * 1024 + col] = f2bf(acc[i][j][r] * 0.125f);
                }
    } else if (z == 1) {
#pragma unroll
        for (int i = 0; i < 4; ++i)
#pragma unroll
            for (int j = 0; j < 4; ++j)
#pragma unroll
                for (int r = 0; r < 8; ++r) {
                    const int row = m0 + 16 * i + rb + r;
                    const int col = n0 + 16 * j + cn;
                    Kb[(long)row * 1024 + col] = f2bf(acc[i][j][r]);
                }
    } else {
#pragma unroll
        for (int i = 0; i < 4; ++i)
#pragma unroll
            for (int j = 0; j < 4; ++j)
#pragma unroll
                for (int r = 0; r < 8; ++r) {
                    const int row  = m0 + 16 * i + rb + r;   // = ((b*3+g)*64+a)*64+j
                    const int col  = n0 + 16 * j + cn;       // = h*64+dd
                    const bf16_t v = f2bf(acc[i][j][r]);
                    const int jj   = row & 63;
                    const int baga = row >> 6;               // (b*3+g)*64+a
                    const int aa   = baga & 63;
                    const int bg   = baga >> 6;              // b*3+g
                    Vt1[((long)baga * 1024 + col) * 64 + jj]           = v;
                    Vt2[(((long)bg * 64 + jj) * 1024 + col) * 64 + aa] = v;
                }
    }
}

// ---------------------------------------------------------------------------
// Kernel 2: s1 logits.  One wave per (b,i,a,h): S1[j,k] = Q[j,:]·Kyz[k,:]
// ---------------------------------------------------------------------------
__global__ __launch_bounds__(256) void k_s1(const bf16_t* __restrict__ Qb,
                                            const bf16_t* __restrict__ Kb,
                                            float* __restrict__ S) {
    const int wid  = threadIdx.x >> 5;
    const int lane = threadIdx.x & 31;
    const int a  = blockIdx.x;
    const int bi = blockIdx.y;                 // b*3+i
    const int b  = bi / 3, i = bi % 3;
    const int h  = blockIdx.z * 8 + wid;
    const int g1 = (i + 1) % 3;

    const bf16_t* A = Qb + ((long)(bi * 64 + a) * 64) * 1024 + h * 64;            // rows j
    const bf16_t* B = Kb + ((long)((b * 3 + g1) * 64 + a) * 64) * 1024 + h * 64;  // rows k

    v8f acc[4][4];
#pragma unroll
    for (int ii = 0; ii < 4; ++ii)
#pragma unroll
        for (int jj = 0; jj < 4; ++jj) acc[ii][jj] = vzero8();

    gemm_64x64_k64(A, 1024, B, 1024, acc);

    float* Sr = S + (((long)(bi * 64 + a) * 64) * 16 + h) * 128;  // row j stride 2048
    const int cn = lane & 15, rb = (lane >> 4) << 3;
#pragma unroll
    for (int ii = 0; ii < 4; ++ii)
#pragma unroll
        for (int jj = 0; jj < 4; ++jj)
#pragma unroll
            for (int r = 0; r < 8; ++r)
                Sr[(long)(16 * ii + rb + r) * 2048 + 16 * jj + cn] = acc[ii][jj][r];
}

// ---------------------------------------------------------------------------
// Kernel 3: s2 logits.  One wave per (b,i,j,h): S2[a,k] = Q[a,:]·Kzx[k,:]
// ---------------------------------------------------------------------------
__global__ __launch_bounds__(256) void k_s2(const bf16_t* __restrict__ Qb,
                                            const bf16_t* __restrict__ Kb,
                                            float* __restrict__ S) {
    const int wid  = threadIdx.x >> 5;
    const int lane = threadIdx.x & 31;
    const int j  = blockIdx.x;
    const int bi = blockIdx.y;
    const int b  = bi / 3, i = bi % 3;
    const int h  = blockIdx.z * 8 + wid;
    const int g2 = (i + 2) % 3;

    const bf16_t* A = Qb + ((long)bi * 4096 + j) * 1024 + h * 64;            // rows a, ld 65536
    const bf16_t* B = Kb + ((long)(b * 3 + g2) * 4096 + j) * 1024 + h * 64;  // rows k, ld 65536

    v8f acc[4][4];
#pragma unroll
    for (int ii = 0; ii < 4; ++ii)
#pragma unroll
        for (int jj = 0; jj < 4; ++jj) acc[ii][jj] = vzero8();

    gemm_64x64_k64(A, 65536, B, 65536, acc);

    float* Sr = S + (((long)bi * 4096 + j) * 16 + h) * 128 + 64;  // row a stride 131072
    const int cn = lane & 15, rb = (lane >> 4) << 3;
#pragma unroll
    for (int ii = 0; ii < 4; ++ii)
#pragma unroll
        for (int jj = 0; jj < 4; ++jj)
#pragma unroll
            for (int r = 0; r < 8; ++r)
                Sr[(long)(16 * ii + rb + r) * 131072 + 16 * jj + cn] = acc[ii][jj][r];
}

// ---------------------------------------------------------------------------
// Kernel 4: softmax over 128 logits, one wave32 per row, bf16 probs out.
// ---------------------------------------------------------------------------
__global__ __launch_bounds__(256) void k_softmax(const float* __restrict__ S,
                                                 bf16_t* __restrict__ P) {
    const int wid  = threadIdx.x >> 5;
    const int lane = threadIdx.x & 31;
    const long row = (long)blockIdx.x * 8 + wid;

    const float4 v = *(const float4*)(S + row * 128 + lane * 4);
    float m = fmaxf(fmaxf(v.x, v.y), fmaxf(v.z, v.w));
#pragma unroll
    for (int off = 16; off > 0; off >>= 1) m = fmaxf(m, __shfl_xor(m, off, 32));

    const float e0 = __expf(v.x - m), e1 = __expf(v.y - m);
    const float e2 = __expf(v.z - m), e3 = __expf(v.w - m);
    float s = e0 + e1 + e2 + e3;
#pragma unroll
    for (int off = 16; off > 0; off >>= 1) s += __shfl_xor(s, off, 32);
    const float inv = 1.0f / s;

    v4bf o;
    o[0] = f2bf(e0 * inv); o[1] = f2bf(e1 * inv);
    o[2] = f2bf(e2 * inv); o[3] = f2bf(e3 * inv);
    *(v4bf*)(P + row * 128 + lane * 4) = o;
}

// ---------------------------------------------------------------------------
// Kernel 5: AV1.  One wave per (b,i,a,h): O[j,dd] = sum_k P1[j,k]*Vt1[dd,k]
// B tile (8KB contiguous) is DMA'd into a private LDS slot by the TDM and
// consumed via ds loads; the DMA overlaps the first A-fragment global loads.
// ---------------------------------------------------------------------------
__global__ __launch_bounds__(256) void k_av1(const bf16_t* __restrict__ P,
                                             const bf16_t* __restrict__ Vt1,
                                             float* __restrict__ O) {
    __shared__ __align__(16) bf16_t vtile[8][4096];
    const int wid  = threadIdx.x >> 5;
    const int lane = threadIdx.x & 31;
    const int a  = blockIdx.x;
    const int bi = blockIdx.y;
    const int b  = bi / 3, i = bi % 3;
    const int h  = blockIdx.z * 8 + wid;
    const int g1 = (i + 1) % 3;

    const bf16_t* A  = P + (((long)(bi * 64 + a) * 64) * 16 + h) * 128;             // rows j, ld 2048
    const bf16_t* Bg = Vt1 + ((long)((b * 3 + g1) * 64 + a) * 1024 + h * 64) * 64;  // 8KB contiguous

    // async DMA of the wave's B tile into LDS (low 32 bits of flat = LDS offset)
    const unsigned lds_off = (unsigned)(unsigned long long)(&vtile[wid][0]);
    tdm_load_8kb(Bg, lds_off);

    v8f acc[4][4];
#pragma unroll
    for (int ii = 0; ii < 4; ++ii)
#pragma unroll
        for (int jj = 0; jj < 4; ++jj) acc[ii][jj] = vzero8();

    // overlap: A fragments for k0=0 while TDM is in flight
    v16bf af[4];
#pragma unroll
    for (int ii = 0; ii < 4; ++ii) af[ii] = load_a_bf16(A, 16 * ii, 2048, 0);

    __builtin_amdgcn_s_wait_tensorcnt(0);

    v16bf bfr[4];
#pragma unroll
    for (int jj = 0; jj < 4; ++jj) bfr[jj] = load_b_lds(&vtile[wid][0], 16 * jj, 0);
#pragma unroll
    for (int ii = 0; ii < 4; ++ii)
#pragma unroll
        for (int jj = 0; jj < 4; ++jj)
            acc[ii][jj] = wmma_bf16(af[ii], bfr[jj], acc[ii][jj]);

#pragma unroll
    for (int ii = 0; ii < 4; ++ii) af[ii] = load_a_bf16(A, 16 * ii, 2048, 32);
#pragma unroll
    for (int jj = 0; jj < 4; ++jj) bfr[jj] = load_b_lds(&vtile[wid][0], 16 * jj, 32);
#pragma unroll
    for (int ii = 0; ii < 4; ++ii)
#pragma unroll
        for (int jj = 0; jj < 4; ++jj)
            acc[ii][jj] = wmma_bf16(af[ii], bfr[jj], acc[ii][jj]);

    float* Or = O + ((long)(bi * 64 + a) * 64) * 1024 + h * 64;   // row j stride 1024
    const int cn = lane & 15, rb = (lane >> 4) << 3;
#pragma unroll
    for (int ii = 0; ii < 4; ++ii)
#pragma unroll
        for (int jj = 0; jj < 4; ++jj)
#pragma unroll
            for (int r = 0; r < 8; ++r)
                Or[(long)(16 * ii + rb + r) * 1024 + 16 * jj + cn] = acc[ii][jj][r];
}

// ---------------------------------------------------------------------------
// Kernel 6: AV2.  One wave per (b,i,j,h): O[a,dd] += sum_k P2[a,k]*Vt2[dd,k]
// Same TDM-staged B tile.  RMW accumulate (ordered after k_av1 on stream).
// ---------------------------------------------------------------------------
__global__ __launch_bounds__(256) void k_av2(const bf16_t* __restrict__ P,
                                             const bf16_t* __restrict__ Vt2,
                                             float* __restrict__ O) {
    __shared__ __align__(16) bf16_t vtile[8][4096];
    const int wid  = threadIdx.x >> 5;
    const int lane = threadIdx.x & 31;
    const int j  = blockIdx.x;
    const int bi = blockIdx.y;
    const int b  = bi / 3, i = bi % 3;
    const int h  = blockIdx.z * 8 + wid;
    const int g2 = (i + 2) % 3;

    const bf16_t* A  = P + (((long)bi * 4096 + j) * 16 + h) * 128 + 64;              // rows a, ld 131072
    const bf16_t* Bg = Vt2 + (((long)(b * 3 + g2) * 64 + j) * 1024 + h * 64) * 64;   // 8KB contiguous

    const unsigned lds_off = (unsigned)(unsigned long long)(&vtile[wid][0]);
    tdm_load_8kb(Bg, lds_off);

    v8f acc[4][4];
#pragma unroll
    for (int ii = 0; ii < 4; ++ii)
#pragma unroll
        for (int jj = 0; jj < 4; ++jj) acc[ii][jj] = vzero8();

    v16bf af[4];
#pragma unroll
    for (int ii = 0; ii < 4; ++ii) af[ii] = load_a_bf16(A, 16 * ii, 131072, 0);

    __builtin_amdgcn_s_wait_tensorcnt(0);

    v16bf bfr[4];
#pragma unroll
    for (int jj = 0; jj < 4; ++jj) bfr[jj] = load_b_lds(&vtile[wid][0], 16 * jj, 0);
#pragma unroll
    for (int ii = 0; ii < 4; ++ii)
#pragma unroll
        for (int jj = 0; jj < 4; ++jj)
            acc[ii][jj] = wmma_bf16(af[ii], bfr[jj], acc[ii][jj]);

#pragma unroll
    for (int ii = 0; ii < 4; ++ii) af[ii] = load_a_bf16(A, 16 * ii, 131072, 32);
#pragma unroll
    for (int jj = 0; jj < 4; ++jj) bfr[jj] = load_b_lds(&vtile[wid][0], 16 * jj, 32);
#pragma unroll
    for (int ii = 0; ii < 4; ++ii)
#pragma unroll
        for (int jj = 0; jj < 4; ++jj)
            acc[ii][jj] = wmma_bf16(af[ii], bfr[jj], acc[ii][jj]);

    float* Or = O + ((long)bi * 4096 + j) * 1024 + h * 64;        // row a stride 65536
    const int cn = lane & 15, rb = (lane >> 4) << 3;
#pragma unroll
    for (int ii = 0; ii < 4; ++ii)
#pragma unroll
        for (int jj = 0; jj < 4; ++jj)
#pragma unroll
            for (int r = 0; r < 8; ++r) {
                float* p = Or + (long)(16 * ii + rb + r) * 65536 + 16 * jj + cn;
                *p += acc[ii][jj][r];
            }
}

// ---------------------------------------------------------------------------
// Kernel 7: output projection  out = Ob @ Wp^T + b  (bf16 A, fp32 out)
// grid = (Nblk=8, Mblk=384) for L2-friendly A reuse.
// ---------------------------------------------------------------------------
__global__ __launch_bounds__(256) void k_proj(const bf16_t* __restrict__ Ob,
                                              const bf16_t* __restrict__ Wp,
                                              const float*  __restrict__ bias,
                                              float* __restrict__ out) {
    const int wid  = threadIdx.x >> 5;
    const int lane = threadIdx.x & 31;
    const int m0 = blockIdx.y * 256 + (wid & 3) * 64;
    const int n0 = blockIdx.x * 128 + (wid >> 2) * 64;

    v8f acc[4][4];
#pragma unroll
    for (int i = 0; i < 4; ++i)
#pragma unroll
        for (int j = 0; j < 4; ++j) acc[i][j] = vzero8();

    for (int k0 = 0; k0 < 1024; k0 += 32) {
        if (k0 + 32 < 1024)
            __builtin_prefetch(Ob + (long)(m0 + (lane & 15)) * 1024 + k0 + 32, 0, 1);
        v16bf af[4], bfr[4];
#pragma unroll
        for (int i = 0; i < 4; ++i) af[i] = load_a_bf16(Ob, m0 + 16 * i, 1024, k0);
#pragma unroll
        for (int j = 0; j < 4; ++j) bfr[j] = load_b_bf16(Wp, n0 + 16 * j, 1024, k0);
#pragma unroll
        for (int i = 0; i < 4; ++i)
#pragma unroll
            for (int j = 0; j < 4; ++j)
                acc[i][j] = wmma_bf16(af[i], bfr[j], acc[i][j]);
    }

    const int cn = lane & 15, rb = (lane >> 4) << 3;
#pragma unroll
    for (int i = 0; i < 4; ++i)
#pragma unroll
        for (int j = 0; j < 4; ++j) {
            const int col = n0 + 16 * j + cn;
            const float bv = bias[col];
#pragma unroll
            for (int r = 0; r < 8; ++r) {
                const int row = m0 + 16 * i + rb + r;
                out[(long)row * 1024 + col] = acc[i][j][r] + bv;
            }
        }
}

// ---------------------------------------------------------------------------
extern "C" void kernel_launch(void* const* d_in, const int* in_sizes, int n_in,
                              void* d_out, int out_size, void* d_ws, size_t ws_size,
                              hipStream_t stream) {
    (void)in_sizes; (void)n_in; (void)out_size; (void)ws_size;

    const float* x    = (const float*)d_in[0];
    const float* wq   = (const float*)d_in[1];
    const float* wk   = (const float*)d_in[2];
    const float* wv   = (const float*)d_in[3];
    const float* wp   = (const float*)d_in[4];
    const float* bias = (const float*)d_in[5];
    float* out = (float*)d_out;

    char* ws = (char*)d_ws;
    // workspace layout (bytes)
    bf16_t* Wb  = (bf16_t*)(ws);                    // 4 x 1Mi bf16      =   8 MiB
    bf16_t* Qb  = (bf16_t*)(ws + 8388608ULL);       // 98304x1024 bf16   = 192 MiB
    bf16_t* Kb  = (bf16_t*)(ws + 209715200ULL);     //                   = 192 MiB
    bf16_t* Vt1 = (bf16_t*)(ws + 411041792ULL);     //                   = 192 MiB
    bf16_t* Vt2 = (bf16_t*)(ws + 612368384ULL);     //                   = 192 MiB
    float*  S   = (float*) (ws + 813694976ULL);     // logits fp32       = 768 MiB
    bf16_t* P   = (bf16_t*)(ws + 1619001344ULL);    // probs bf16        = 384 MiB
    float*  Obf = (float*) (ws + 813694976ULL);     // O aliases dead S
    bf16_t* Ob  = (bf16_t*)(ws + 1619001344ULL);    // Ob aliases dead P
    bf16_t* Xb  = (bf16_t*)(ws + 2021654528ULL);    // bf16 x            = 192 MiB

    const dim3 blk(256);

    // 0) weights and x -> bf16
    k_cvt_w<<<dim3(4096, 1, 4), blk, 0, stream>>>(wq, wk, wv, wp, Wb);
    k_cvt4 <<<dim3(98304), blk, 0, stream>>>(x, Xb);

    // 1) QKV projection (z = Q/K/V)
    k_qkv<<<dim3(8, 384, 3), blk, 0, stream>>>(Xb, Wb, Qb, Kb, Vt1, Vt2);

    // 2) logits along j-axis and a-axis
    k_s1<<<dim3(64, 24, 2), blk, 0, stream>>>(Qb, Kb, S);
    k_s2<<<dim3(64, 24, 2), blk, 0, stream>>>(Qb, Kb, S);

    // 3) softmax over concatenated 128 logits
    k_softmax<<<dim3(196608), blk, 0, stream>>>(S, P);

    // 4) attention * V (two halves, TDM-staged B tiles)
    k_av1<<<dim3(64, 24, 2), blk, 0, stream>>>(P, Vt1, Obf);
    k_av2<<<dim3(64, 24, 2), blk, 0, stream>>>(P, Vt2, Obf);

    // 5) O -> bf16, then output projection + bias
    k_cvt4 <<<dim3(98304), blk, 0, stream>>>(Obf, Ob);
    k_proj<<<dim3(8, 384), blk, 0, stream>>>(Ob, Wb + 3 * 1048576, bias, out);
}
